// AttentionHead_33981781245964
// MI455X (gfx1250) — compile-verified, matrix operand order checked by reading, not run on hardware
//
#include <hip/hip_runtime.h>

#define DEV static __device__ __forceinline__

constexpr int B_ = 4, T_ = 4096, C_ = 2048, H_ = 128;

typedef __attribute__((ext_vector_type(16))) __bf16          v16bf;
typedef __attribute__((ext_vector_type(8)))  float           v8f;
typedef __attribute__((ext_vector_type(16))) unsigned short  us16;
typedef __attribute__((ext_vector_type(4)))  unsigned int    u32x4;
typedef __attribute__((ext_vector_type(8)))  int             i32x8;
typedef __attribute__((ext_vector_type(4)))  int             i32x4;

#if __has_builtin(__builtin_amdgcn_tensor_load_to_lds)
#define USE_TDM 1
#else
#define USE_TDM 0
#endif

DEV unsigned short f2bf(float f) {
  unsigned u = __builtin_bit_cast(unsigned, f);
  u += 0x7FFFu + ((u >> 16) & 1u);           // round-to-nearest-even
  return (unsigned short)(u >> 16);
}

DEV v16bf ldb16(const unsigned short* p) {   // 32B aligned by construction
  return __builtin_bit_cast(v16bf, *(const us16*)p);
}

// ---------------------------------------------------------------------------
// TDM: DMA a 2-D bf16 tile (tile1 rows x tile0 elems, row stride stride0 elems)
// from global into LDS at byte offset ldsAddr. Descriptor per ISA 08 §8.3/8.4.
// tensor_dim == tile_dim so the OOB path can never trigger.
// ---------------------------------------------------------------------------
DEV void tdm_load_2d(const void* gsrc, unsigned ldsAddr, int tile0, int tile1,
                     int stride0) {
#if USE_TDM
  unsigned long long ga = (unsigned long long)gsrc;
  u32x4 g0;
  g0[0] = 1u;                                    // count=1 (valid), user mode
  g0[1] = ldsAddr;                               // D#.lds_addr (bytes)
  g0[2] = (unsigned)ga;                          // global_addr[31:0]
  g0[3] = (unsigned)((ga >> 32) & 0x01FFFFFFu)   // global_addr[56:32]
        | (2u << 30);                            // type=2 ("image")
  i32x8 g1;
  g1[0] = 1 << 16;                               // data_size=1 (2B), wg_mask=0
  g1[1] = (tile0 & 0xFFFF) << 16;                // tensor_dim0[15:0] @ bits 63:48
  g1[2] = (tile1 & 0xFFFF) << 16;                // tensor_dim1[15:0] @ bits 95:80
  g1[3] = (tile0 & 0xFFFF) << 16;                // tile_dim0 @ bits 127:112
  g1[4] = tile1 & 0xFFFF;                        // tile_dim1 (tile_dim2=0)
  g1[5] = stride0;                               // tensor_dim0_stride[31:0]
  g1[6] = 0;                                     // stride hi / dim1_stride = 0
  g1[7] = 0;
  i32x4 gz4 = {0, 0, 0, 0};
  i32x8 gz8 = {0, 0, 0, 0, 0, 0, 0, 0};
  // 6-arg form (clang-23 / therock-10.0 headers): groups 0..4 + cpol
  __builtin_amdgcn_tensor_load_to_lds(g0, g1, gz4, gz4, gz8, 0);
#else
  (void)gsrc; (void)ldsAddr; (void)tile0; (void)tile1; (void)stride0;
#endif
}

DEV unsigned lds_off(const void* p) { return (unsigned)(unsigned long long)p; }

// ---------------------------------------------------------------------------
// Kernel 0: W [C][H] fp32  ->  WT [H][C] bf16
// ---------------------------------------------------------------------------
__global__ void cvt_w_kernel(const float* __restrict__ W, unsigned short* __restrict__ WT) {
  int idx = blockIdx.x * blockDim.x + threadIdx.x;
  if (idx < C_ * H_) {
    int h = idx / C_;
    int c = idx - h * C_;
    WT[idx] = f2bf(W[(size_t)c * H_ + h]);
  }
}

// ---------------------------------------------------------------------------
// Kernel 1: projection  out = x @ W + b  (WMMA bf16, fp32 accum)
// ---------------------------------------------------------------------------
__global__ __launch_bounds__(256) void qkv_proj_kernel(
    const float* __restrict__ x, const unsigned short* __restrict__ WT,
    const float* __restrict__ bias, unsigned short* __restrict__ out, int colMajor) {
  const int lane    = threadIdx.x & 31;
  const int wave    = threadIdx.x >> 5;
  const int rowBase = blockIdx.x * 128 + wave * 16;
  const int mrow    = lane & 15;
  const int khalf   = (lane >> 4) << 4;
  const int rHi     = (lane >> 4) << 3;

  v8f acc[8];
#pragma unroll
  for (int i = 0; i < 8; ++i)
#pragma unroll
    for (int e = 0; e < 8; ++e) acc[i][e] = 0.0f;

  for (int kk = 0; kk < C_; kk += 32) {
    const float* xp = x + (size_t)(rowBase + mrow) * C_ + kk + khalf;
    __builtin_prefetch(xp + 64, 0, 0);          // global_prefetch a line ahead
    us16 au;
#pragma unroll
    for (int i = 0; i < 4; ++i) {
      float4 t = *(const float4*)(xp + 4 * i);
      au[4 * i + 0] = f2bf(t.x); au[4 * i + 1] = f2bf(t.y);
      au[4 * i + 2] = f2bf(t.z); au[4 * i + 3] = f2bf(t.w);
    }
    v16bf a = __builtin_bit_cast(v16bf, au);
#pragma unroll
    for (int nt = 0; nt < 8; ++nt) {
      v16bf b = ldb16(WT + (size_t)(nt * 16 + mrow) * C_ + kk + khalf);
      acc[nt] = __builtin_amdgcn_wmma_f32_16x16x32_bf16(
          false, a, false, b, (short)0, acc[nt], false, false);
    }
  }

#pragma unroll
  for (int nt = 0; nt < 8; ++nt) {
    const int   c  = nt * 16 + mrow;
    const float bv = bias[c];
#pragma unroll
    for (int j = 0; j < 8; ++j) {
      const int r = rowBase + j + rHi;
      const unsigned short h = f2bf(acc[nt][j] + bv);
      if (!colMajor) {
        out[(size_t)r * H_ + c] = h;
      } else {
        const int b  = r >> 12;          // r / T_
        const int tt = r & (T_ - 1);
        out[((size_t)(b * H_ + c)) * T_ + tt] = h;
      }
    }
  }
}

// ---------------------------------------------------------------------------
// Kernel 2: causal flash attention. k/v tiles staged block-wide in LDS by the
// Tensor Data Mover (double-buffered, TENSORcnt-pipelined); 8 waves compute
// 16 query rows each with bf16 WMMA + fp32 online softmax.
// ---------------------------------------------------------------------------
__global__ __launch_bounds__(256) void flash_attn_kernel(
    const unsigned short* __restrict__ qb, const unsigned short* __restrict__ kb,
    const unsigned short* __restrict__ vT, float* __restrict__ out) {
  __shared__ alignas(32) unsigned short kTile[2][32 * H_];   // [s 0..31][h]
  __shared__ alignas(32) unsigned short vTile[2][H_ * 32];   // [h][s 0..31]
  __shared__ alignas(32) unsigned short pLds[8][16 * 32];

  const int tid   = threadIdx.x;
  const int lane  = tid & 31;
  const int wave  = tid >> 5;
  const int bId   = blockIdx.x >> 5;
  const int qTile = blockIdx.x & 31;
  const int tq0   = qTile * 128 + wave * 16;
  const int mrow  = lane & 15;
  const int khalf = (lane >> 4) << 4;
  const int rHi   = (lane >> 4) << 3;

  const unsigned short* kBatch = kb + (size_t)bId * T_ * H_;
  const unsigned short* vBatch = vT + (size_t)bId * H_ * T_;

  v16bf qA[4];
#pragma unroll
  for (int hk = 0; hk < 4; ++hk)
    qA[hk] = ldb16(qb + ((size_t)(bId * T_ + tq0 + mrow)) * H_ + hk * 32 + khalf);

  v8f   O[8];
  float mSt[8], lSt[8];
#pragma unroll
  for (int i = 0; i < 8; ++i) {
    mSt[i] = -__builtin_inff(); lSt[i] = 0.0f;
#pragma unroll
    for (int e = 0; e < 8; ++e) O[i][e] = 0.0f;
  }

  const float scale = 0.022097086912079608f;   // 2048^-0.5 (reference C**-0.5)
  const int   nIter = 4 * (qTile + 1);         // uniform across block (causal)

#if USE_TDM
  if (wave == 0) {                             // prologue: DMA tile 0 -> buf 0
    tdm_load_2d(kBatch, lds_off(kTile[0]), H_, 32, H_);
    tdm_load_2d(vBatch, lds_off(vTile[0]), 32, H_, T_);
  }
#endif

  for (int i = 0; i < nIter; ++i) {
    const int s0 = i * 32;
    __syncthreads();                           // readers of buf[(i+1)&1] done
#if USE_TDM
    if (wave == 0) {
      if (i + 1 < nIter) {                     // issue next tile, then wait for
        const int sn = s0 + 32;                // current (in-order completion)
        tdm_load_2d(kBatch + (size_t)sn * H_, lds_off(kTile[(i + 1) & 1]), H_, 32, H_);
        tdm_load_2d(vBatch + sn,              lds_off(vTile[(i + 1) & 1]), 32, H_, T_);
        __builtin_amdgcn_s_wait_tensorcnt(2);
      } else {
        __builtin_amdgcn_s_wait_tensorcnt(0);
      }
    }
#else
    {                                          // cooperative-copy fallback
      uint4* kd = (uint4*)kTile[i & 1];
      const uint4* ks = (const uint4*)(kBatch + (size_t)s0 * H_);
      for (int c = tid; c < 32 * H_ / 8; c += 256) kd[c] = ks[c];
      for (int c = tid; c < H_ * 4; c += 256) {
        int r = c >> 2, cc = c & 3;
        ((uint4*)vTile[i & 1])[c] =
            *(const uint4*)(vBatch + (size_t)r * T_ + s0 + cc * 8);
      }
    }
#endif
    __syncthreads();                           // tile i visible to all waves
    const unsigned short* kT  = kTile[i & 1];
    const unsigned short* vTl = vTile[i & 1];

    // ---- S = q @ k^T : two 16x16 tiles (cols s0..s0+31), K over H from LDS
    v8f sAcc[2];
#pragma unroll
    for (int st = 0; st < 2; ++st) {
#pragma unroll
      for (int e = 0; e < 8; ++e) sAcc[st][e] = 0.0f;
#pragma unroll
      for (int hk = 0; hk < 4; ++hk) {
        v16bf bm = ldb16(kT + (st * 16 + mrow) * H_ + hk * 32 + khalf);
        sAcc[st] = __builtin_amdgcn_wmma_f32_16x16x32_bf16(
            false, qA[hk], false, bm, (short)0, sAcc[st], false, false);
      }
    }

    // ---- scale, causal mask, online softmax update
#pragma unroll
    for (int j = 0; j < 8; ++j) {
      const int qr = tq0 + j + rHi;
#pragma unroll
      for (int st = 0; st < 2; ++st) {
        const int   c  = s0 + st * 16 + mrow;
        const float sv = sAcc[st][j] * scale;
        sAcc[st][j] = (c <= qr) ? sv : -__builtin_inff();
      }
      float v = fmaxf(sAcc[0][j], sAcc[1][j]);
      v = fmaxf(v, __shfl_xor(v, 1, 32));
      v = fmaxf(v, __shfl_xor(v, 2, 32));
      v = fmaxf(v, __shfl_xor(v, 4, 32));
      v = fmaxf(v, __shfl_xor(v, 8, 32));
      const float mNew  = fmaxf(mSt[j], v);    // finite after block 0
      const float alpha = __expf(mSt[j] - mNew);
      const float p0    = __expf(sAcc[0][j] - mNew);
      const float p1    = __expf(sAcc[1][j] - mNew);
      sAcc[0][j] = p0; sAcc[1][j] = p1;
      float rs = p0 + p1;
      rs += __shfl_xor(rs, 1, 32);
      rs += __shfl_xor(rs, 2, 32);
      rs += __shfl_xor(rs, 4, 32);
      rs += __shfl_xor(rs, 8, 32);
      lSt[j] = lSt[j] * alpha + rs;
      mSt[j] = mNew;
#pragma unroll
      for (int nt = 0; nt < 8; ++nt) O[nt][j] *= alpha;
    }

    // ---- P: D-layout -> A-layout via LDS (intra-wave)
#pragma unroll
    for (int j = 0; j < 8; ++j) {
      const int pr = j + rHi;
      pLds[wave][pr * 32 + mrow]      = f2bf(sAcc[0][j]);
      pLds[wave][pr * 32 + 16 + mrow] = f2bf(sAcc[1][j]);
    }
    __builtin_amdgcn_wave_barrier();
    asm volatile("s_wait_dscnt 0" ::: "memory");
    v16bf pA = __builtin_bit_cast(v16bf, *(const us16*)&pLds[wave][mrow * 32 + khalf]);

    // ---- O += P @ v (v tile in LDS, rows = h, contiguous in s)
#pragma unroll
    for (int nt = 0; nt < 8; ++nt) {
      v16bf bm = ldb16(vTl + (nt * 16 + mrow) * 32 + khalf);
      O[nt] = __builtin_amdgcn_wmma_f32_16x16x32_bf16(
          false, pA, false, bm, (short)0, O[nt], false, false);
    }
    __builtin_amdgcn_wave_barrier();
  }

  // ---- normalize and store fp32 [B,T,H]
#pragma unroll
  for (int j = 0; j < 8; ++j) {
    const float inv = 1.0f / lSt[j];
    const int   r   = bId * T_ + tq0 + j + rHi;
#pragma unroll
    for (int nt = 0; nt < 8; ++nt)
      out[(size_t)r * H_ + nt * 16 + mrow] = O[nt][j] * inv;
  }
}

// ---------------------------------------------------------------------------
extern "C" void kernel_launch(void* const* d_in, const int* in_sizes, int n_in,
                              void* d_out, int out_size, void* d_ws, size_t ws_size,
                              hipStream_t stream) {
  (void)in_sizes; (void)n_in; (void)out_size; (void)ws_size;
  const float* x  = (const float*)d_in[0];
  const float* Wk = (const float*)d_in[1];
  const float* bk = (const float*)d_in[2];
  const float* Wq = (const float*)d_in[3];
  const float* bq = (const float*)d_in[4];
  const float* Wv = (const float*)d_in[5];
  const float* bv = (const float*)d_in[6];
  float* out = (float*)d_out;

  char* ws = (char*)d_ws;
  const size_t whSz = (size_t)C_ * H_ * sizeof(unsigned short);        // 512 KB
  const size_t qkSz = (size_t)B_ * T_ * H_ * sizeof(unsigned short);   // 4 MB
  unsigned short* WqT = (unsigned short*)(ws);
  unsigned short* WkT = (unsigned short*)(ws + whSz);
  unsigned short* WvT = (unsigned short*)(ws + 2 * whSz);
  unsigned short* qB  = (unsigned short*)(ws + 3 * whSz);
  unsigned short* kB  = (unsigned short*)(ws + 3 * whSz + qkSz);
  unsigned short* vTB = (unsigned short*)(ws + 3 * whSz + 2 * qkSz);

  const int cvtBlocks = (C_ * H_ + 255) / 256;
  cvt_w_kernel<<<cvtBlocks, 256, 0, stream>>>(Wq, WqT);
  cvt_w_kernel<<<cvtBlocks, 256, 0, stream>>>(Wk, WkT);
  cvt_w_kernel<<<cvtBlocks, 256, 0, stream>>>(Wv, WvT);

  const int projBlocks = (B_ * T_) / 128;   // 128
  qkv_proj_kernel<<<projBlocks, 256, 0, stream>>>(x, WqT, bq, qB, 0);
  qkv_proj_kernel<<<projBlocks, 256, 0, stream>>>(x, WkT, bk, kB, 0);
  qkv_proj_kernel<<<projBlocks, 256, 0, stream>>>(x, WvT, bv, vTB, 1);

  flash_attn_kernel<<<B_ * (T_ / 128), 256, 0, stream>>>(qB, kB, vTB, out);
}
